// GNNModel_30142080483538
// MI455X (gfx1250) — matrix operand order repaired and checked by reading, not compiled, not run on hardware
//
#include <hip/hip_runtime.h>
#include <hip/hip_bf16.h>

typedef __attribute__((ext_vector_type(16))) _Float16 v16h;
typedef __attribute__((ext_vector_type(8)))  float    v8f;

#define C64 64

// ---------------------------------------------------------------- utilities

__global__ void fill0_kernel(float* __restrict__ p, long long n) {
    long long i = (long long)blockIdx.x * blockDim.x + threadIdx.x;
    long long stride = (long long)gridDim.x * blockDim.x;
    for (; i < n; i += stride) p[i] = 0.0f;
}

__global__ void count_nodes_kernel(const int* __restrict__ batch, float* __restrict__ cnt, int n) {
    int i = blockIdx.x * blockDim.x + threadIdx.x;
    if (i < n) atomicAdd(&cnt[batch[i]], 1.0f);
}

// ---------------------------------------------------------------- GIN aggregation

// First conv: features are [N,6]
__global__ void gin1_agg_kernel(const float* __restrict__ x,
                                const int* __restrict__ src, const int* __restrict__ dst,
                                float* __restrict__ agg6, int E) {
    int e = blockIdx.x * blockDim.x + threadIdx.x;
    if (e >= E) return;
    int s = src[e], d = dst[e];
    #pragma unroll
    for (int c = 0; c < 6; ++c)
        atomicAdd(&agg6[(long long)d * 6 + c], x[(long long)s * 6 + c]);
}

// Second conv: features are [N,64]; one thread per (edge, 4-channel group)
__global__ void gin2_agg_kernel(const float* __restrict__ xb,
                                const int* __restrict__ src, const int* __restrict__ dst,
                                float* __restrict__ agg, int E) {
    long long t = (long long)blockIdx.x * blockDim.x + threadIdx.x;
    if (t >= (long long)E * 16) return;
    int e = (int)(t >> 4);
    int q = ((int)t & 15) * 4;
    int s = src[e], d = dst[e];
    const float4 v = *(const float4*)(xb + (long long)s * C64 + q);
    float* o = agg + (long long)d * C64 + q;
    atomicAdd(o + 0, v.x);
    atomicAdd(o + 1, v.y);
    atomicAdd(o + 2, v.z);
    atomicAdd(o + 3, v.w);
}

// ---------------------------------------------------------------- weight fragment prep
// Pack W[k][64] (row-major) into B-fragment order:
//   frag[((ng*nkc + kc)*32 + lane)*16 + e]  holds  W[kc*32 + (lane>>4)*16 + e][ng*16 + (lane&15)]
__global__ void prep_frag_kernel(const float* __restrict__ W, _Float16* __restrict__ F,
                                 int Krows, int nkc) {
    int idx = blockIdx.x * blockDim.x + threadIdx.x;
    int total = 4 * nkc * 32 * 16;
    if (idx >= total) return;
    int e    = idx & 15;
    int t    = idx >> 4;
    int lane = t & 31; t >>= 5;
    int kc   = t % nkc;
    int ng   = t / nkc;
    int k = kc * 32 + (lane >> 4) * 16 + e;      // e = j*2+pos -> k runs 0..15 within half
    int n = ng * 16 + (lane & 15);
    float v = (k < Krows) ? W[k * C64 + n] : 0.0f;
    F[idx] = (_Float16)v;
}

// ---------------------------------------------------------------- WMMA node MLP
// out[16 rows x 64] = relu(X @ Wa + ba) @ Wb + bb   (relu only on inner layer)
// MODE 0: X = x[N,6] + agg6[N,6]  (K padded to 32 with zeros)
// MODE 1: X = xin[N,64] + agg[N,64]
template <int MODE>
__global__ void __launch_bounds__(128) mlp_pair_wmma(
    const float* __restrict__ xin, const float* __restrict__ agg,
    float* __restrict__ out,
    const _Float16* __restrict__ WaF, const float* __restrict__ ba,
    const _Float16* __restrict__ WbF, const float* __restrict__ bb,
    int nNodes) {
    __shared__ float    Xs[16][65];   // padded stride to spread LDS banks
    __shared__ _Float16 Hs[16][66];

    const int tid  = threadIdx.x;
    const int wid  = tid >> 5;        // 4 waves -> 4 column groups of 16
    const int lane = tid & 31;
    const int m    = lane & 15;
    const int half = lane >> 4;
    const int r0   = blockIdx.x * 16;

    if (MODE == 0) {
        for (int i = tid; i < 16 * 32; i += 128) Xs[i >> 5][i & 31] = 0.0f;
        __syncthreads();
        if (tid < 96) {
            int r = tid / 6, c = tid % 6;
            int row = r0 + r;
            float v = 0.0f;
            if (row < nNodes) v = xin[(long long)row * 6 + c] + agg[(long long)row * 6 + c];
            Xs[r][c] = v;
        }
    } else {
        int r  = tid >> 3;
        int c0 = (tid & 7) * 8;
        int row = r0 + r;
        #pragma unroll
        for (int c = 0; c < 8; ++c) {
            float v = 0.0f;
            if (row < nNodes) v = xin[(long long)row * C64 + c0 + c] + agg[(long long)row * C64 + c0 + c];
            Xs[r][c0 + c] = v;
        }
    }
    __syncthreads();

    // A-fragment K mapping (16-bit A, 16x32): element e -> K = half*16 + e (matches B packing)
    int kl[16];
    #pragma unroll
    for (int e = 0; e < 16; ++e) {
        kl[e] = (half << 4) + e;
    }

    const int ch = wid * 16 + m;      // output channel this lane owns in D

    // ---- layer 1: X @ Wa + ba, ReLU, into LDS (f16)
    constexpr int NKCA = (MODE == 0) ? 1 : 2;
    v8f acc = {0.f, 0.f, 0.f, 0.f, 0.f, 0.f, 0.f, 0.f};
    const v16h* WaV = (const v16h*)WaF;
    #pragma unroll
    for (int kc = 0; kc < NKCA; ++kc) {
        v16h a;
        #pragma unroll
        for (int e = 0; e < 16; ++e) a[e] = (_Float16)Xs[m][kc * 32 + kl[e]];
        v16h b = WaV[(wid * NKCA + kc) * 32 + lane];
        acc = __builtin_amdgcn_wmma_f32_16x16x32_f16(false, a, false, b, (short)0, acc,
                                                     false, false);
    }
    {
        float bav = ba[ch];
        #pragma unroll
        for (int r = 0; r < 8; ++r) {
            float v = acc[r] + bav;
            Hs[half * 8 + r][ch] = (_Float16)(v > 0.0f ? v : 0.0f);
        }
    }
    __syncthreads();

    // ---- layer 2: H @ Wb + bb -> global
    v8f acc2 = {0.f, 0.f, 0.f, 0.f, 0.f, 0.f, 0.f, 0.f};
    const v16h* WbV = (const v16h*)WbF;
    #pragma unroll
    for (int kc = 0; kc < 2; ++kc) {
        v16h a;
        #pragma unroll
        for (int e = 0; e < 16; ++e) a[e] = Hs[m][kc * 32 + kl[e]];
        v16h b = WbV[(wid * 2 + kc) * 32 + lane];
        acc2 = __builtin_amdgcn_wmma_f32_16x16x32_f16(false, a, false, b, (short)0, acc2,
                                                      false, false);
    }
    {
        float bbv = bb[ch];
        #pragma unroll
        for (int r = 0; r < 8; ++r) {
            int row = r0 + half * 8 + r;
            if (row < nNodes) out[(long long)row * C64 + ch] = acc2[r] + bbv;
        }
    }
}

// ---------------------------------------------------------------- graph layernorm

__global__ void ln_stats_kernel(const float* __restrict__ x, const int* __restrict__ batch,
                                float* __restrict__ sum, float* __restrict__ sumsq, int n) {
    int i = blockIdx.x * blockDim.x + threadIdx.x;
    if (i >= n) return;
    const float4* p = (const float4*)(x + (long long)i * C64);
    float s = 0.f, ss = 0.f;
    #pragma unroll
    for (int q = 0; q < 16; ++q) {
        float4 v = p[q];
        s  += v.x + v.y + v.z + v.w;
        ss += v.x * v.x + v.y * v.y + v.z * v.z + v.w * v.w;
    }
    int b = batch[i];
    atomicAdd(&sum[b], s);
    atomicAdd(&sumsq[b], ss);
}

// out = relu((x - mean) * rsqrt(var+eps) * w + b)
__global__ void ln_apply_kernel(const float* __restrict__ x, const int* __restrict__ batch,
                                const float* __restrict__ cnt, const float* __restrict__ sum,
                                const float* __restrict__ sumsq,
                                const float* __restrict__ w, const float* __restrict__ bia,
                                float* __restrict__ out, int n) {
    long long t = (long long)blockIdx.x * blockDim.x + threadIdx.x;
    if (t >= (long long)n * 16) return;
    int i = (int)(t >> 4);
    int q = ((int)t & 15) * 4;
    int b = batch[i];
    float c = cnt[b];
    float norm = (c > 1.0f ? c : 1.0f) * (float)C64;
    float mean = sum[b] / norm;
    float var  = sumsq[b] / norm - mean * mean;
    float inv  = __frsqrt_rn(var + 1e-5f);
    float4 v = *(const float4*)(x + (long long)i * C64 + q);
    float4 wv = *(const float4*)(w + q);
    float4 bv = *(const float4*)(bia + q);
    float4 o;
    o.x = (v.x - mean) * inv * wv.x + bv.x; o.x = o.x > 0.f ? o.x : 0.f;
    o.y = (v.y - mean) * inv * wv.y + bv.y; o.y = o.y > 0.f ? o.y : 0.f;
    o.z = (v.z - mean) * inv * wv.z + bv.z; o.z = o.z > 0.f ? o.z : 0.f;
    o.w = (v.w - mean) * inv * wv.w + bv.w; o.w = o.w > 0.f ? o.w : 0.f;
    *(float4*)(out + (long long)i * C64 + q) = o;
}

// ---------------------------------------------------------------- pooling

__global__ void pool_kernel(const float* __restrict__ x, const int* __restrict__ batch,
                            float* __restrict__ emb, int n) {
    long long t = (long long)blockIdx.x * blockDim.x + threadIdx.x;
    if (t >= (long long)n * 16) return;
    int i = (int)(t >> 4);
    int q = ((int)t & 15) * 4;
    int b = batch[i];
    float4 v = *(const float4*)(x + (long long)i * C64 + q);
    float* o = emb + (long long)b * C64 + q;
    atomicAdd(o + 0, v.x);
    atomicAdd(o + 1, v.y);
    atomicAdd(o + 2, v.z);
    atomicAdd(o + 3, v.w);
}

__global__ void pool_fin_kernel(float* __restrict__ emb, const float* __restrict__ cnt, int G) {
    int idx = blockIdx.x * blockDim.x + threadIdx.x;
    if (idx >= G * C64) return;
    int g = idx >> 6;
    float c = cnt[g];
    float add = emb[idx];
    emb[idx] = add + add / (c > 1.0f ? c : 1.0f);   // add-pool + mean-pool
}

// ---------------------------------------------------------------- head

__global__ void head1_kernel(const float* __restrict__ e1, const float* __restrict__ e2,
                             const float* __restrict__ d1, const float* __restrict__ d2,
                             const float* __restrict__ Wf1, const float* __restrict__ bf1,
                             float* __restrict__ h1, int G) {
    int idx = blockIdx.x * blockDim.x + threadIdx.x;
    if (idx >= G * 128) return;
    int g = idx >> 7, o = idx & 127;
    float acc = bf1[o];
    for (int k = 0; k < 64; ++k) acc += e1[g * 64 + k] * Wf1[k * 128 + o];
    for (int k = 0; k < 64; ++k) acc += e2[g * 64 + k] * Wf1[(64 + k) * 128 + o];
    for (int k = 0; k < 5;  ++k) acc += d1[g * 5 + k] * Wf1[(128 + k) * 128 + o];
    for (int k = 0; k < 5;  ++k) acc += d2[g * 5 + k] * Wf1[(133 + k) * 128 + o];
    h1[idx] = acc > 0.f ? acc : 0.f;
}

__global__ void head2_kernel(const float* __restrict__ h1, const float* __restrict__ Wf2,
                             const float* __restrict__ bf2, float* __restrict__ h2, int G) {
    int idx = blockIdx.x * blockDim.x + threadIdx.x;
    if (idx >= G * 64) return;
    int g = idx >> 6, o = idx & 63;
    float acc = bf2[o];
    for (int k = 0; k < 128; ++k) acc += h1[g * 128 + k] * Wf2[k * 64 + o];
    h2[idx] = acc > 0.f ? acc : 0.f;
}

__global__ void head3_kernel(const float* __restrict__ h2, const float* __restrict__ Wo,
                             const float* __restrict__ bo, float* __restrict__ out, int G) {
    int g = blockIdx.x * blockDim.x + threadIdx.x;
    if (g >= G) return;
    float acc = bo[0];
    for (int k = 0; k < 64; ++k) acc += h2[g * 64 + k] * Wo[k];
    out[g] = acc;
}

// ---------------------------------------------------------------- host side

static inline int cdiv(long long a, int b) { return (int)((a + b - 1) / b); }

static void run_graph_tower(const float* x, const int* ei, const int* batch,
                            int N, int E, int G,
                            float* bufA, float* bufB, float* agg6, float* stats,
                            float* emb,
                            const _Float16* W1f, const float* b1,
                            const _Float16* W2f, const float* b2,
                            const float* ln1w, const float* ln1b,
                            const _Float16* W3f, const float* b3,
                            const _Float16* W4f, const float* b4,
                            const float* ln2w, const float* ln2b,
                            hipStream_t stream) {
    const int* src = ei;
    const int* dst = ei + E;
    float* cnt = stats;
    float* sum = stats + G;
    float* ssq = stats + 2 * G;
    const int tiles = cdiv(N, 16);

    fill0_kernel<<<cdiv((long long)N * 6, 256), 256, 0, stream>>>(agg6, (long long)N * 6);
    fill0_kernel<<<cdiv(G, 256), 256, 0, stream>>>(cnt, G);
    count_nodes_kernel<<<cdiv(N, 256), 256, 0, stream>>>(batch, cnt, N);

    // GIN conv 1
    gin1_agg_kernel<<<cdiv(E, 256), 256, 0, stream>>>(x, src, dst, agg6, E);
    mlp_pair_wmma<0><<<tiles, 128, 0, stream>>>(x, agg6, bufA, W1f, b1, W2f, b2, N);

    // LayerNorm(graph) + ReLU -> bufB
    fill0_kernel<<<cdiv(2 * G, 256), 256, 0, stream>>>(sum, 2 * G);
    ln_stats_kernel<<<cdiv(N, 256), 256, 0, stream>>>(bufA, batch, sum, ssq, N);
    ln_apply_kernel<<<cdiv((long long)N * 16, 256), 256, 0, stream>>>(
        bufA, batch, cnt, sum, ssq, ln1w, ln1b, bufB, N);

    // GIN conv 2 (agg lives in bufA, then MLP writes bufA in place)
    fill0_kernel<<<cdiv((long long)N * 64, 256), 256, 0, stream>>>(bufA, (long long)N * 64);
    gin2_agg_kernel<<<cdiv((long long)E * 16, 256), 256, 0, stream>>>(bufB, src, dst, bufA, E);
    mlp_pair_wmma<1><<<tiles, 128, 0, stream>>>(bufB, bufA, bufA, W3f, b3, W4f, b4, N);

    // LayerNorm(graph) + ReLU -> bufB
    fill0_kernel<<<cdiv(2 * G, 256), 256, 0, stream>>>(sum, 2 * G);
    ln_stats_kernel<<<cdiv(N, 256), 256, 0, stream>>>(bufA, batch, sum, ssq, N);
    ln_apply_kernel<<<cdiv((long long)N * 16, 256), 256, 0, stream>>>(
        bufA, batch, cnt, sum, ssq, ln2w, ln2b, bufB, N);

    // add-pool + mean-pool
    fill0_kernel<<<cdiv(G * 64, 256), 256, 0, stream>>>(emb, G * 64);
    pool_kernel<<<cdiv((long long)N * 16, 256), 256, 0, stream>>>(bufB, batch, emb, N);
    pool_fin_kernel<<<cdiv(G * 64, 256), 256, 0, stream>>>(emb, cnt, G);
}

extern "C" void kernel_launch(void* const* d_in, const int* in_sizes, int n_in,
                              void* d_out, int out_size, void* d_ws, size_t ws_size,
                              hipStream_t stream) {
    const int N = in_sizes[0] / 6;
    const int E = in_sizes[1] / 2;
    const int G = in_sizes[6] / 5;

    const float* x1  = (const float*)d_in[0];
    const int*   ei1 = (const int*)d_in[1];
    const int*   bt1 = (const int*)d_in[2];
    const float* x2  = (const float*)d_in[3];
    const int*   ei2 = (const int*)d_in[4];
    const int*   bt2 = (const int*)d_in[5];
    const float* d1  = (const float*)d_in[6];
    const float* d2  = (const float*)d_in[7];
    const float* W1  = (const float*)d_in[8];
    const float* b1  = (const float*)d_in[9];
    const float* W2  = (const float*)d_in[10];
    const float* b2  = (const float*)d_in[11];
    const float* ln1w = (const float*)d_in[12];
    const float* ln1b = (const float*)d_in[13];
    const float* W3  = (const float*)d_in[14];
    const float* b3  = (const float*)d_in[15];
    const float* W4  = (const float*)d_in[16];
    const float* b4  = (const float*)d_in[17];
    const float* ln2w = (const float*)d_in[18];
    const float* ln2b = (const float*)d_in[19];
    const float* Wf1 = (const float*)d_in[20];
    const float* bf1 = (const float*)d_in[21];
    const float* Wf2 = (const float*)d_in[22];
    const float* bf2 = (const float*)d_in[23];
    const float* Wo  = (const float*)d_in[24];
    const float* bo  = (const float*)d_in[25];
    float* out = (float*)d_out;

    // ---- carve workspace
    char* base = (char*)d_ws;
    size_t off = 0;
    auto carve = [&](size_t bytes) -> void* {
        void* p = base + off;
        off = (off + bytes + 255) & ~(size_t)255;
        return p;
    };
    float* bufA  = (float*)carve((size_t)N * 64 * 4);
    float* bufB  = (float*)carve((size_t)N * 64 * 4);
    float* agg6  = (float*)carve((size_t)N * 6 * 4);
    float* stats = (float*)carve((size_t)3 * G * 4);
    float* emb1  = (float*)carve((size_t)G * 64 * 4);
    float* emb2  = (float*)carve((size_t)G * 64 * 4);
    float* h1    = (float*)carve((size_t)G * 128 * 4);
    float* h2    = (float*)carve((size_t)G * 64 * 4);
    _Float16* W1f = (_Float16*)carve(4 * 1 * 32 * 16 * 2);
    _Float16* W2f = (_Float16*)carve(4 * 2 * 32 * 16 * 2);
    _Float16* W3f = (_Float16*)carve(4 * 2 * 32 * 16 * 2);
    _Float16* W4f = (_Float16*)carve(4 * 2 * 32 * 16 * 2);

    // ---- weight fragments (f16, WMMA B layout)
    prep_frag_kernel<<<cdiv(4 * 1 * 512, 256), 256, 0, stream>>>(W1, W1f, 6, 1);
    prep_frag_kernel<<<cdiv(4 * 2 * 512, 256), 256, 0, stream>>>(W2, W2f, 64, 2);
    prep_frag_kernel<<<cdiv(4 * 2 * 512, 256), 256, 0, stream>>>(W3, W3f, 64, 2);
    prep_frag_kernel<<<cdiv(4 * 2 * 512, 256), 256, 0, stream>>>(W4, W4f, 64, 2);

    // ---- two towers (shared weights)
    run_graph_tower(x1, ei1, bt1, N, E, G, bufA, bufB, agg6, stats, emb1,
                    W1f, b1, W2f, b2, ln1w, ln1b, W3f, b3, W4f, b4, ln2w, ln2b, stream);
    run_graph_tower(x2, ei2, bt2, N, E, G, bufA, bufB, agg6, stats, emb2,
                    W1f, b1, W2f, b2, ln1w, ln1b, W3f, b3, W4f, b4, ln2w, ln2b, stream);

    // ---- head
    head1_kernel<<<cdiv(G * 128, 256), 256, 0, stream>>>(emb1, emb2, d1, d2, Wf1, bf1, h1, G);
    head2_kernel<<<cdiv(G * 64, 256), 256, 0, stream>>>(h1, Wf2, bf2, h2, G);
    head3_kernel<<<cdiv(G, 256), 256, 0, stream>>>(h2, Wo, bo, out, G);
}
